// EGNN_38749194944928
// MI455X (gfx1250) — matrix operand order repaired and checked by reading, not compiled, not run on hardware
//
#include <hip/hip_runtime.h>
#include <hip/hip_bf16.h>

// ---------------------------------------------------------------------------
// EGNN forward for MI455X (gfx1250): wave32, bf16 WMMA (f32 accum), TDM
// (tensor_load_to_lds) staging of the edge-feature stream.
// ---------------------------------------------------------------------------

typedef __attribute__((ext_vector_type(16))) __bf16 v16bf;
typedef __attribute__((ext_vector_type(8)))  __bf16 v8bf;
typedef __attribute__((ext_vector_type(8)))  float  v8f;
typedef __attribute__((ext_vector_type(4)))  unsigned int v4u;
typedef __attribute__((ext_vector_type(4)))  int    v4i;
typedef __attribute__((ext_vector_type(8)))  int    v8i;

#define DEVFN __device__ __forceinline__

#if __has_builtin(__builtin_amdgcn_tensor_load_to_lds) && __has_builtin(__builtin_amdgcn_s_wait_tensorcnt)
#define HAS_TDM 1
#else
#define HAS_TDM 0
#endif

constexpr int kN    = 40000;
constexpr int kE    = 512000;
constexpr int kDT   = 128;
constexpr int kH    = 8;
constexpr int kDH   = 16;
constexpr int kNG   = 64;
constexpr int kNGEN = 10000;

// ---------------------------------------------------------------------------
// WMMA fragment helpers (CDNA5 ISA 7.12.2, wave32)
// A 16x32 bf16: lanes 0-15 row M=lane&15, K={0..7,16..23}; lanes 16-31 same
// rows, K={8..15,24..31}.
// ---------------------------------------------------------------------------

DEVFN v16bf a_frag_global(const float* __restrict__ A, int lda, int m0, int k0) {
  int lane = threadIdx.x & 31;
  const float* p = A + (size_t)(m0 + (lane & 15)) * lda + (k0 + ((lane >> 4) << 3));
  v16bf a;
#pragma unroll
  for (int j = 0; j < 8; ++j) { a[j] = (__bf16)p[j]; a[j + 8] = (__bf16)p[j + 16]; }
  return a;
}

// A fragment from an LDS-resident f32 chunk [128 rows x 32 cols], row stride 32
DEVFN v16bf a_frag_lds_f32(const float* S, int m0) {
  int lane = threadIdx.x & 31;
  const float* p = S + (m0 + (lane & 15)) * 32 + ((lane >> 4) << 3);
  v16bf a;
#pragma unroll
  for (int j = 0; j < 8; ++j) { a[j] = (__bf16)p[j]; a[j + 8] = (__bf16)p[j + 16]; }
  return a;
}

DEVFN v16bf a_frag_lds_bf16(const __bf16* S, int m0, int k0) {
  int lane = threadIdx.x & 31;
  const __bf16* p = S + (m0 + (lane & 15)) * kDT + (k0 + ((lane >> 4) << 3));
  v8bf lo = *(const v8bf*)p;           // 16B aligned -> ds_load_b128
  v8bf hi = *(const v8bf*)(p + 16);
  return __builtin_shufflevector(lo, hi, 0, 1, 2, 3, 4, 5, 6, 7,
                                 8, 9, 10, 11, 12, 13, 14, 15);
}

// B 32x16 bf16 pre-swizzled: per (k-block of 32, n-tile of 16) each lane holds
// 16 contiguous bf16 (K = (lane>>4)*8 + (j&7) + (j>>3)*16, N = lane&15).
DEVFN v16bf b_frag(const __bf16* __restrict__ Wswz, int k0, int nt) {
  int lane = threadIdx.x & 31;
  const __bf16* p = Wswz + ((((k0 >> 5) << 3) + nt) << 9) + lane * 16;
  v8bf lo = *(const v8bf*)p;           // 32B aligned -> 2x global_load_b128
  v8bf hi = *(const v8bf*)(p + 8);
  return __builtin_shufflevector(lo, hi, 0, 1, 2, 3, 4, 5, 6, 7,
                                 8, 9, 10, 11, 12, 13, 14, 15);
}

DEVFN v8f wmma_bf16(v16bf a, v16bf b, v8f c) {
  return __builtin_amdgcn_wmma_f32_16x16x32_bf16(false, a, false, b, (short)0, c,
                                                 false, false);
}

// ---------------------------------------------------------------------------
// TDM: 2D f32 tile (rows x cols, row stride strideElems) global -> LDS.
// D# bit layout per CDNA5 ISA §8.3/8.4. LDS byte offset = low 32 bits of the
// flat shared-aperture address (ISA §10.2).
// ---------------------------------------------------------------------------
#if HAS_TDM
DEVFN void tdm_load_f32_tile(const float* g, const void* ldsPtr,
                             int cols, int rows, int strideElems) {
  unsigned long long ga = (unsigned long long)(uintptr_t)g;
  unsigned ldsOff = (unsigned)(uintptr_t)ldsPtr;
  v4u g0;
  g0[0] = 1u;                                          // count=1 valid D#
  g0[1] = ldsOff;                                      // lds_addr [63:32]
  g0[2] = (unsigned)(ga & 0xFFFFFFFFu);                // global_addr lo
  g0[3] = (unsigned)((ga >> 32) & 0x01FFFFFFu) | (2u << 30);  // addr hi | type=2
  v8i g1;
  g1[0] = (2 << 16);                                   // data_size=4B, mask=0
  g1[1] = (cols & 0xFFFF) << 16;                       // tensor_dim0[15:0]
  g1[2] = ((unsigned)cols >> 16) | ((rows & 0xFFFF) << 16);   // td0 hi | td1 lo
  g1[3] = ((unsigned)rows >> 16) | ((cols & 0xFFFF) << 16);   // td1 hi | tile_dim0
  g1[4] = rows & 0xFFFF;                               // tile_dim1 (tile_dim2=0)
  g1[5] = strideElems;                                 // tensor_dim0_stride lo32
  g1[6] = 0;
  g1[7] = 0;
  v4i z4 = {};
#if __clang_major__ >= 23
  v8i z8 = {};
  __builtin_amdgcn_tensor_load_to_lds(g0, g1, z4, z4, z8, 0);
#else
  __builtin_amdgcn_tensor_load_to_lds(g0, g1, z4, z4, 0);
#endif
}
#endif

// ---------------------------------------------------------------------------
// Weight swizzle: f32 row-major (K=128 x N=128) -> bf16 B-fragment order
// ---------------------------------------------------------------------------
__global__ void swizzle_w(const float* __restrict__ W, __bf16* __restrict__ dst) {
  int idx = blockIdx.x * blockDim.x + threadIdx.x;
  if (idx >= 16384) return;
  int j = idx & 15, lane = (idx >> 4) & 31, tile = idx >> 9;
  int nt = tile & 7, kb = tile >> 3;
  int k = kb * 32 + ((lane >> 4) << 3) + (j & 7) + ((j >> 3) << 4);
  int n = nt * 16 + (lane & 15);
  dst[idx] = (__bf16)W[k * 128 + n];
}

// gw (384x128): fold concat GEMM into two 128x128 GEMMs.
__global__ void swizzle_gate(const float* __restrict__ gw,
                             __bf16* __restrict__ dA, __bf16* __restrict__ dB) {
  int idx = blockIdx.x * blockDim.x + threadIdx.x;
  if (idx >= 16384) return;
  int j = idx & 15, lane = (idx >> 4) & 31, tile = idx >> 9;
  int nt = tile & 7, kb = tile >> 3;
  int k = kb * 32 + ((lane >> 4) << 3) + (j & 7) + ((j >> 3) << 4);
  int n = nt * 16 + (lane & 15);
  float w0 = gw[k * 128 + n];
  float w1 = gw[(128 + k) * 128 + n];
  float w2 = gw[(256 + k) * 128 + n];
  dA[idx] = (__bf16)(w0 + w2);
  dB[idx] = (__bf16)(w1 - w2);
}

// ---------------------------------------------------------------------------
// Generic GEMM: C[M x 128] = act(A[M x 128] @ W (+ A2 @ W2) + bias)
// 256 thr = 8 waves; each wave owns a 32x128 strip (2 A frags, 16 WMMA tiles)
// so every B fragment feeds two WMMAs. ACT: 0 none, 1 lrelu(0.01), 2 sigmoid.
// ---------------------------------------------------------------------------
template <int ACT, bool DUAL>
__global__ void gemm128_kernel(const float* __restrict__ A, const __bf16* __restrict__ W,
                               const float* __restrict__ A2, const __bf16* __restrict__ W2,
                               const float* __restrict__ bias, float* __restrict__ C, int M) {
  int wave = threadIdx.x >> 5, lane = threadIdx.x & 31;
  int m0 = (blockIdx.x * 8 + wave) * 32;
  if (m0 >= M) return;  // uniform per wave: EXEC stays all-ones for WMMA
  v8f acc[2][8] = {};
#pragma unroll
  for (int k0 = 0; k0 < kDT; k0 += 32) {
    v16bf a0 = a_frag_global(A, kDT, m0, k0);
    v16bf a1 = a_frag_global(A, kDT, m0 + 16, k0);
#pragma unroll
    for (int nt = 0; nt < 8; ++nt) {
      v16bf b = b_frag(W, k0, nt);
      acc[0][nt] = wmma_bf16(a0, b, acc[0][nt]);
      acc[1][nt] = wmma_bf16(a1, b, acc[1][nt]);
    }
  }
  if constexpr (DUAL) {
#pragma unroll
    for (int k0 = 0; k0 < kDT; k0 += 32) {
      v16bf a0 = a_frag_global(A2, kDT, m0, k0);
      v16bf a1 = a_frag_global(A2, kDT, m0 + 16, k0);
#pragma unroll
      for (int nt = 0; nt < 8; ++nt) {
        v16bf b = b_frag(W2, k0, nt);
        acc[0][nt] = wmma_bf16(a0, b, acc[0][nt]);
        acc[1][nt] = wmma_bf16(a1, b, acc[1][nt]);
      }
    }
  }
  int nCol = lane & 15, mAdd = (lane >> 4) << 3;
#pragma unroll
  for (int s = 0; s < 2; ++s) {
#pragma unroll
    for (int nt = 0; nt < 8; ++nt) {
      int col = nt * 16 + nCol;
      float bv = bias[col];
#pragma unroll
      for (int r = 0; r < 8; ++r) {
        float v = acc[s][nt][r] + bv;
        if (ACT == 1) v = v > 0.f ? v : 0.01f * v;
        else if (ACT == 2) v = 1.f / (1.f + __expf(-v));
        C[(size_t)(m0 + s * 16 + mAdd + r) * kDT + col] = v;
      }
    }
  }
}

// ---------------------------------------------------------------------------
// Fused edge kernel: 128 edges / block (256 thr, 8 waves).
//   edge_s chunk staged via TDM (tensor_load_to_lds) 16KB at a time
//   hidden = lrelu(edge_s @ m1w[:128] + d_ij*m1w[128] + m1b)    (GEMM1 -> LDS)
//   m_ij   = hidden @ m2w + m2b                                 (GEMM2)
//   a_ij   = q[src] * k[dst] * m_ij / 4                         (-> LDS, bf16)
//   edge_s_new = a_ij @ ew + eb                                 (GEMM3 -> out)
//   scores = sum |a_ij| per head -> global + atomic max per (src,h)
//   att MLP on a_ij -> masked dx scatter into dagg
// ---------------------------------------------------------------------------
__global__ void edge_kernel(
    const float* __restrict__ edge_s, const int* __restrict__ eidx,
    const float* __restrict__ pos, const unsigned char* __restrict__ maskinv,
    const float* __restrict__ qbuf, const float* __restrict__ kbuf,
    const __bf16* __restrict__ Wm1, const __bf16* __restrict__ Wm2,
    const __bf16* __restrict__ Wew,
    const float* __restrict__ m1w_full, const float* __restrict__ m1b,
    const float* __restrict__ m2b, const float* __restrict__ eb,
    const float* __restrict__ att1w, const float* __restrict__ att1b,
    const float* __restrict__ att2w, const float* __restrict__ att2b,
    const float* __restrict__ headw, const int* __restrict__ proPtr,
    float* __restrict__ scoresOut, unsigned int* __restrict__ nmaxBits,
    float* __restrict__ edge_out, float* __restrict__ dagg) {
  __shared__ float  sh_es[128 * 32];   // TDM-staged edge_s k-chunk (16KB)
  __shared__ __bf16 sh_h[128 * kDT];   // hidden, then a_ij (reused, 32KB)
  __shared__ float sh_d[128];          // d_ij (>=0) or -1 when masked
  __shared__ float sh_dx[128][3];      // raw pos diff
  __shared__ int   sh_src[128], sh_dst[128];
  __shared__ float sh_m1last[kDT];

  const int tid = threadIdx.x;
  const int e0 = blockIdx.x * 128;

  if (tid < 128) {
    int e = e0 + tid;
    int s = eidx[e], d = eidx[kE + e];
    sh_src[tid] = s; sh_dst[tid] = d;
    float d0 = pos[s * 3 + 0] - pos[d * 3 + 0];
    float d1 = pos[s * 3 + 1] - pos[d * 3 + 1];
    float d2 = pos[s * 3 + 2] - pos[d * 3 + 2];
    sh_dx[tid][0] = d0; sh_dx[tid][1] = d1; sh_dx[tid][2] = d2;
    float a0 = d0 + 1e-6f, a1 = d1 + 1e-6f, a2 = d2 + 1e-6f;
    float n1 = sqrtf(a0 * a0 + a1 * a1 + a2 * a2);
    sh_d[tid] = maskinv[e] ? -1.0f : n1 * 0.1f;
    sh_m1last[tid] = m1w_full[128 * kDT + tid];  // last row of (129,128) m1w
  }
  __syncthreads();

  const int wave = tid >> 5, lane = tid & 31;
  const int mloc = wave * 16;
  const int nCol = lane & 15, mAdd = (lane >> 4) << 3;

  // ---- GEMM1: hidden, A staged chunk-wise through LDS by the TDM ----
  v8f acc[8] = {};
  for (int k0 = 0; k0 < kDT; k0 += 32) {
#if HAS_TDM
    if (wave == 0) {  // one wave programs the DMA; TDM ignores EXEC
      tdm_load_f32_tile(edge_s + (size_t)e0 * kDT + k0, sh_es, 32, 128, kDT);
      __builtin_amdgcn_s_wait_tensorcnt(0);
    }
#else
    for (int t = tid; t < 128 * 32; t += 256)
      sh_es[t] = edge_s[(size_t)e0 * kDT + (size_t)(t >> 5) * kDT + k0 + (t & 31)];
#endif
    __syncthreads();
    v16bf a = a_frag_lds_f32(sh_es, mloc);
#pragma unroll
    for (int nt = 0; nt < 8; ++nt) acc[nt] = wmma_bf16(a, b_frag(Wm1, k0, nt), acc[nt]);
    __syncthreads();  // chunk consumed before next TDM overwrite
  }
#pragma unroll
  for (int nt = 0; nt < 8; ++nt) {
    int col = nt * 16 + nCol;
    float bv = m1b[col], wl = sh_m1last[col];
#pragma unroll
    for (int r = 0; r < 8; ++r) {
      int m = mloc + mAdd + r;
      float v = acc[nt][r] + bv + sh_d[m] * wl;
      v = v > 0.f ? v : 0.01f * v;
      sh_h[m * kDT + col] = (__bf16)v;
    }
  }
  __syncthreads();

  // ---- GEMM2: m_ij ----
  {
    v8f z = {};
#pragma unroll
    for (int nt = 0; nt < 8; ++nt) acc[nt] = z;
  }
#pragma unroll
  for (int k0 = 0; k0 < kDT; k0 += 32) {
    v16bf a = a_frag_lds_bf16(sh_h, mloc, k0);
#pragma unroll
    for (int nt = 0; nt < 8; ++nt) acc[nt] = wmma_bf16(a, b_frag(Wm2, k0, nt), acc[nt]);
  }
  __syncthreads();  // hidden reads done before overwriting sh_h with a_ij

  // ---- a_ij = q[src]*k[dst]*m_ij/sqrt(DH) -> LDS (q/k gathers hit L2) ----
#pragma unroll
  for (int nt = 0; nt < 8; ++nt) {
    int col = nt * 16 + nCol;
    float bv = m2b[col];
#pragma unroll
    for (int r = 0; r < 8; ++r) {
      int m = mloc + mAdd + r;
      float mij = acc[nt][r] + bv;
      float kv = kbuf[(size_t)sh_dst[m] * kDT + col];
      float qv = qbuf[(size_t)sh_src[m] * kDT + col];
      sh_h[m * kDT + col] = (__bf16)(qv * kv * mij * 0.25f);
    }
  }
  __syncthreads();

  // ---- GEMM3: edge_s_new = a_ij @ ew + eb ----
  {
    v8f z = {};
#pragma unroll
    for (int nt = 0; nt < 8; ++nt) acc[nt] = z;
  }
#pragma unroll
  for (int k0 = 0; k0 < kDT; k0 += 32) {
    v16bf a = a_frag_lds_bf16(sh_h, mloc, k0);
#pragma unroll
    for (int nt = 0; nt < 8; ++nt) acc[nt] = wmma_bf16(a, b_frag(Wew, k0, nt), acc[nt]);
  }
#pragma unroll
  for (int nt = 0; nt < 8; ++nt) {
    int col = nt * 16 + nCol;
    float bv = eb[col];
#pragma unroll
    for (int r = 0; r < 8; ++r)
      edge_out[(size_t)(e0 + mloc + mAdd + r) * kDT + col] = acc[nt][r] + bv;
  }

  // ---- scores, atomic max, att MLP, masked dx scatter (one edge/thread) ----
  if (tid < 128) {
    int e = e0 + tid, s = sh_src[tid];
    float att = 0.f;
#pragma unroll
    for (int h = 0; h < kH; ++h) {
      float ah[kDH];
      float sc = 0.f;
#pragma unroll
      for (int k = 0; k < kDH; ++k) {
        float x = (float)sh_h[tid * kDT + h * kDH + k];
        ah[k] = x;
        sc += fabsf(x);
      }
      scoresOut[(size_t)e * kH + h] = sc;
      atomicMax(&nmaxBits[s * kH + h], __float_as_uint(sc));  // sc >= 0
      float a2 = att2b[0];
#pragma unroll
      for (int j = 0; j < kDH / 2; ++j) {
        float t = att1b[j];
#pragma unroll
        for (int k = 0; k < kDH; ++k) t += ah[k] * att1w[k * (kDH / 2) + j];
        t = t > 0.f ? t : 0.01f * t;
        a2 += t * att2w[j];
      }
      att += a2 * headw[h];
    }
    int pro = *proPtr;
    if (s >= pro && sh_d[tid] >= 0.f) {
      float d0 = sh_dx[tid][0], d1 = sh_dx[tid][1], d2 = sh_dx[tid][2];
      float inv = att / (sqrtf(d0 * d0 + d1 * d1 + d2 * d2) + 1e-6f);
      atomicAdd(&dagg[s * 3 + 0], d0 * inv);
      atomicAdd(&dagg[s * 3 + 1], d1 * inv);
      atomicAdd(&dagg[s * 3 + 2], d2 * inv);
    }
  }
}

// ---------------------------------------------------------------------------
// Segment softmax passes + weighted aggregation
// ---------------------------------------------------------------------------
__global__ void softmax_ex(const int* __restrict__ eidx, float* __restrict__ scores,
                           const unsigned int* __restrict__ nmaxBits,
                           float* __restrict__ nden) {
  int i = blockIdx.x * blockDim.x + threadIdx.x;  // E*H
  if (i >= kE * kH) return;
  int e = i >> 3, h = i & 7;
  int s = eidx[e];
  float mx = __uint_as_float(nmaxBits[s * kH + h]);
  float ex = __expf(scores[i] - mx);
  scores[i] = ex;
  atomicAdd(&nden[s * kH + h], ex);
}

__global__ void aggregate(const int* __restrict__ eidx, const float* __restrict__ ex,
                          const float* __restrict__ nden, const float* __restrict__ vbuf,
                          float* __restrict__ agg) {
  int t = blockIdx.x * blockDim.x + threadIdx.x;  // E*4, 32 channels each
  if (t >= kE * 4) return;
  int e = t >> 2, c0 = (t & 3) * 32;
  int s = eidx[e], d = eidx[kE + e];
#pragma unroll
  for (int hh = 0; hh < 2; ++hh) {
    int h = (c0 >> 4) + hh;
    float w = ex[(size_t)e * kH + h] / (nden[s * kH + h] + 1e-16f);
#pragma unroll
    for (int k = 0; k < kDH; ++k) {
      int c = h * kDH + k;
      atomicAdd(&agg[(size_t)s * kDT + c], w * vbuf[(size_t)d * kDT + c]);
    }
  }
}

// ---------------------------------------------------------------------------
// GraphNorm (3 passes over 64 groups) + elementwise helpers
// ---------------------------------------------------------------------------
__global__ void gn_pass1(const float* __restrict__ x, const int* __restrict__ batch,
                         float* __restrict__ gsum, float* __restrict__ gcnt) {
  int i = blockIdx.x * blockDim.x + threadIdx.x;
  if (i >= kN * kDT) return;
  int row = i >> 7, c = i & 127;
  int b = batch[row];
  atomicAdd(&gsum[b * kDT + c], x[i]);
  if (c == 0) atomicAdd(&gcnt[b], 1.0f);
}

__global__ void gn_pass2(const float* __restrict__ x, const int* __restrict__ batch,
                         const float* __restrict__ gsum, const float* __restrict__ gcnt,
                         const float* __restrict__ ms, float* __restrict__ xc,
                         float* __restrict__ gvar) {
  int i = blockIdx.x * blockDim.x + threadIdx.x;
  if (i >= kN * kDT) return;
  int row = i >> 7, c = i & 127;
  int b = batch[row];
  float o = x[i] - ms[c] * (gsum[b * kDT + c] / gcnt[b]);
  xc[i] = o;
  atomicAdd(&gvar[b * kDT + c], o * o);
}

__global__ void gn_pass3(const float* __restrict__ xc, const int* __restrict__ batch,
                         const float* __restrict__ gvar, const float* __restrict__ gcnt,
                         const float* __restrict__ w, const float* __restrict__ bb,
                         float* __restrict__ out) {
  int i = blockIdx.x * blockDim.x + threadIdx.x;
  if (i >= kN * kDT) return;
  int row = i >> 7, c = i & 127;
  int b = batch[row];
  float var = gvar[b * kDT + c] / gcnt[b];
  out[i] = w[c] * xc[i] * rsqrtf(var + 1e-5f) + bb[c];
}

__global__ void fma_gate(const float* __restrict__ g, const float* __restrict__ a,
                         const float* __restrict__ b, float* __restrict__ out) {
  int i = blockIdx.x * blockDim.x + threadIdx.x;
  if (i >= kN * kDT) return;
  out[i] = g[i] * a[i] + b[i];
}

__global__ void pos_out_kernel(const float* __restrict__ pos, const int* __restrict__ gidx,
                               const float* __restrict__ dagg, float* __restrict__ out) {
  int i = blockIdx.x * blockDim.x + threadIdx.x;
  if (i >= kNGEN * 3) return;
  int r = i / 3, c = i - r * 3;
  int n = gidx[r];
  out[i] = pos[n * 3 + c] + dagg[n * 3 + c];
}

// ---------------------------------------------------------------------------
// Host launcher
// ---------------------------------------------------------------------------
extern "C" void kernel_launch(void* const* d_in, const int* in_sizes, int n_in,
                              void* d_out, int out_size, void* d_ws, size_t ws_size,
                              hipStream_t stream) {
  (void)in_sizes; (void)n_in; (void)out_size; (void)ws_size;
  const float* node_s   = (const float*)d_in[0];
  const float* edge_s   = (const float*)d_in[1];
  const int*   eidx     = (const int*)d_in[2];
  const float* pos      = (const float*)d_in[4];
  const int*   gidx     = (const int*)d_in[6];
  const unsigned char* maskinv = (const unsigned char*)d_in[7];
  const int*   proPtr   = (const int*)d_in[8];
  const int*   batch    = (const int*)d_in[9];
  const float* qw  = (const float*)d_in[10]; const float* qb  = (const float*)d_in[11];
  const float* kw  = (const float*)d_in[12]; const float* kb  = (const float*)d_in[13];
  const float* vw  = (const float*)d_in[14]; const float* vb  = (const float*)d_in[15];
  const float* m1w = (const float*)d_in[16]; const float* m1b = (const float*)d_in[17];
  const float* m2w = (const float*)d_in[18]; const float* m2b = (const float*)d_in[19];
  const float* m2fw= (const float*)d_in[20]; const float* m2fb= (const float*)d_in[21];
  const float* ew  = (const float*)d_in[22]; const float* eb  = (const float*)d_in[23];
  const float* gw  = (const float*)d_in[24]; const float* gb  = (const float*)d_in[25];
  const float* f1w = (const float*)d_in[26]; const float* f1b = (const float*)d_in[27];
  const float* f2w = (const float*)d_in[28]; const float* f2b = (const float*)d_in[29];
  const float* ln1w= (const float*)d_in[30]; const float* ln1b= (const float*)d_in[31];
  const float* ln1m= (const float*)d_in[32];
  const float* ln2w= (const float*)d_in[33]; const float* ln2b= (const float*)d_in[34];
  const float* ln2m= (const float*)d_in[35];
  const float* a1w = (const float*)d_in[36]; const float* a1b = (const float*)d_in[37];
  const float* a2w = (const float*)d_in[38]; const float* a2b = (const float*)d_in[39];
  const float* hdw = (const float*)d_in[40];

  size_t cur = 0;
  char* wsBase = (char*)d_ws;
  auto carve = [&](size_t bytes) -> char* {
    char* p = wsBase + cur;
    cur += (bytes + 255) & ~(size_t)255;
    return p;
  };
  const size_t WSZ = 16384 * sizeof(__bf16);
  __bf16* Wq   = (__bf16*)carve(WSZ);
  __bf16* Wk   = (__bf16*)carve(WSZ);
  __bf16* Wv   = (__bf16*)carve(WSZ);
  __bf16* Wm1  = (__bf16*)carve(WSZ);
  __bf16* Wm2  = (__bf16*)carve(WSZ);
  __bf16* Wm2f = (__bf16*)carve(WSZ);
  __bf16* Wew  = (__bf16*)carve(WSZ);
  __bf16* Wf1  = (__bf16*)carve(WSZ);
  __bf16* Wf2  = (__bf16*)carve(WSZ);
  __bf16* Wga  = (__bf16*)carve(WSZ);
  __bf16* Wgb  = (__bf16*)carve(WSZ);
  const size_t NB = (size_t)kN * kDT * sizeof(float);
  float* B0 = (float*)carve(NB);  // q  -> fin hidden
  float* B1 = (float*)carve(NB);  // k  -> fin
  float* B2 = (float*)carve(NB);  // v  -> centered (xc)
  float* B3 = (float*)carve(NB);  // agg -> h1
  float* B4 = (float*)carve(NB);  // ns_new -> x2
  float* B5 = (float*)carve(NB);  // g
  float* B6 = (float*)carve(NB);  // x1
  float* scores = (float*)carve((size_t)kE * kH * sizeof(float));
  unsigned int* nmax = (unsigned int*)carve((size_t)kN * kH * sizeof(unsigned int));
  float* nden = (float*)carve((size_t)kN * kH * sizeof(float));
  float* dagg = (float*)carve((size_t)kN * 3 * sizeof(float));
  float* gsum = (float*)carve((size_t)(kNG * kDT * 2 + kNG) * sizeof(float));
  float* gcnt = gsum + kNG * kDT;
  float* gvar = gcnt + kNG;

  float* out_h    = (float*)d_out;
  float* out_edge = out_h + (size_t)kN * kDT;
  float* out_pos  = out_edge + (size_t)kE * kDT;

  // weight prep
  swizzle_w<<<64, 256, 0, stream>>>(qw,  Wq);
  swizzle_w<<<64, 256, 0, stream>>>(kw,  Wk);
  swizzle_w<<<64, 256, 0, stream>>>(vw,  Wv);
  swizzle_w<<<64, 256, 0, stream>>>(m1w, Wm1);   // first 128 rows of (129,128)
  swizzle_w<<<64, 256, 0, stream>>>(m2w, Wm2);
  swizzle_w<<<64, 256, 0, stream>>>(m2fw, Wm2f);
  swizzle_w<<<64, 256, 0, stream>>>(ew,  Wew);
  swizzle_w<<<64, 256, 0, stream>>>(f1w, Wf1);
  swizzle_w<<<64, 256, 0, stream>>>(f2w, Wf2);
  swizzle_gate<<<64, 256, 0, stream>>>(gw, Wga, Wgb);

  // zero accumulators
  hipMemsetAsync(B3,   0, NB, stream);
  hipMemsetAsync(nmax, 0, (size_t)kN * kH * sizeof(unsigned int), stream);
  hipMemsetAsync(nden, 0, (size_t)kN * kH * sizeof(float), stream);
  hipMemsetAsync(dagg, 0, (size_t)kN * 3 * sizeof(float), stream);

  const int gN = (kN + 255) / 256;         // gemm blocks (256 rows/block)
  const int gEw = (kN * kDT + 255) / 256;  // elementwise over N*DT

  // q, k, v
  gemm128_kernel<0, false><<<gN, 256, 0, stream>>>(node_s, Wq, nullptr, nullptr, qb, B0, kN);
  gemm128_kernel<0, false><<<gN, 256, 0, stream>>>(node_s, Wk, nullptr, nullptr, kb, B1, kN);
  gemm128_kernel<0, false><<<gN, 256, 0, stream>>>(node_s, Wv, nullptr, nullptr, vb, B2, kN);

  // fused edge pipeline
  edge_kernel<<<kE / 128, 256, 0, stream>>>(
      edge_s, eidx, pos, maskinv, B0, B1, Wm1, Wm2, Wew, m1w, m1b, m2b, eb,
      a1w, a1b, a2w, a2b, hdw, proPtr, scores, nmax, out_edge, dagg);

  // segment softmax + aggregation
  softmax_ex<<<(kE * kH + 255) / 256, 256, 0, stream>>>(eidx, scores, nmax, nden);
  aggregate<<<(kE * 4 + 255) / 256, 256, 0, stream>>>(eidx, scores, nden, B2, B3);

  // node-side MLPs
  gemm128_kernel<0, false><<<gN, 256, 0, stream>>>(B3, Wm2f, nullptr, nullptr, m2fb, B4, kN);
  gemm128_kernel<2, true ><<<gN, 256, 0, stream>>>(B4, Wga, node_s, Wgb, gb, B5, kN);
  fma_gate<<<gEw, 256, 0, stream>>>(B5, B4, node_s, B6);   // x1 = g*ns_new + node_s

  // graph norm 1 -> h1 (B3)
  hipMemsetAsync(gsum, 0, (size_t)(kNG * kDT * 2 + kNG) * sizeof(float), stream);
  gn_pass1<<<gEw, 256, 0, stream>>>(B6, batch, gsum, gcnt);
  gn_pass2<<<gEw, 256, 0, stream>>>(B6, batch, gsum, gcnt, ln1m, B2, gvar);
  gn_pass3<<<gEw, 256, 0, stream>>>(B2, batch, gvar, gcnt, ln1w, ln1b, B3);

  // fin = lrelu(h1@f1+b) @ f2 + b
  gemm128_kernel<1, false><<<gN, 256, 0, stream>>>(B3, Wf1, nullptr, nullptr, f1b, B0, kN);
  gemm128_kernel<0, false><<<gN, 256, 0, stream>>>(B0, Wf2, nullptr, nullptr, f2b, B1, kN);
  fma_gate<<<gEw, 256, 0, stream>>>(B5, B1, B3, B4);       // x2 = g*fin + h1

  // graph norm 2 -> final h (d_out)
  hipMemsetAsync(gsum, 0, (size_t)(kNG * kDT * 2 + kNG) * sizeof(float), stream);
  gn_pass1<<<gEw, 256, 0, stream>>>(B4, batch, gsum, gcnt);
  gn_pass2<<<gEw, 256, 0, stream>>>(B4, batch, gsum, gcnt, ln2m, B2, gvar);
  gn_pass3<<<gEw, 256, 0, stream>>>(B2, batch, gvar, gcnt, ln2w, ln2b, out_h);

  // pos_out
  pos_out_kernel<<<(kNGEN * 3 + 255) / 256, 256, 0, stream>>>(pos, gidx, dagg, out_pos);
}
